// PIGNN_42734924595226
// MI455X (gfx1250) — matrix operand order repaired and checked by reading, not compiled, not run on hardware
//
#include <hip/hip_runtime.h>
#include <hip/hip_bf16.h>

#define NN 50000
#define NE 800000
#define HD 128
#define NLAY 6
#define WPB 4   // waves per block (128 threads, wave32)

typedef __bf16 bf16;
typedef __attribute__((ext_vector_type(16))) bf16 v16bf;
typedef __attribute__((ext_vector_type(8)))  bf16 v8bf;
typedef __attribute__((ext_vector_type(8)))  float v8f;

__device__ __forceinline__ int laneid() { return (int)(threadIdx.x & 31); }

// native f32 -> bf16 (RNE) — single v_cvt op on gfx1250
__device__ __forceinline__ bf16 f2bf(float f) { return (bf16)f; }

// ---- WMMA operand loaders (wave32, 16x16x32 bf16) ----
// A (16xK tile, bf16 row-major): lane hi-half owns K+8 group.
// element e: K = kb + (e<8 ? e : e+8), kb = (lane>=16)*8
__device__ __forceinline__ v16bf load_a_g(const bf16* p /* row base + k0 */) {
    int kb = (laneid() >> 4) << 3;
    v8bf lo = *(const v8bf*)(p + kb);
    v8bf hi = *(const v8bf*)(p + kb + 16);
    v16bf r;
#pragma unroll
    for (int e = 0; e < 8; ++e) { r[e] = lo[e]; r[e + 8] = hi[e]; }
    return r;
}
// A from f32 source (converted on the fly)
__device__ __forceinline__ v16bf load_a_f32(const float* p) {
    int kb = (laneid() >> 4) << 3;
    v16bf r;
#pragma unroll
    for (int e = 0; e < 8; ++e) {
        r[e]     = f2bf(p[kb + e]);
        r[e + 8] = f2bf(p[kb + 16 + e]);
    }
    return r;
}
// A from f32 source, K padded with zeros beyond kreal (first MLP layers)
__device__ __forceinline__ v16bf load_a_pad(const float* row, int kreal) {
    int kb = (laneid() >> 4) << 3;
    v16bf r;
#pragma unroll
    for (int e = 0; e < 16; ++e) {
        int k = kb + (e < 8 ? e : e + 8);
        r[e] = f2bf((k < kreal) ? row[k] : 0.0f);
    }
    return r;
}
// B (KxN tile) from transposed weights Wt[N][K] (bf16 row-major):
// element e: K = kb + (e<8 ? e : e+8), kb = (lane>=16)*8; N = lane&15
__device__ __forceinline__ v16bf load_b(const bf16* Wt, int ld, int n0, int k0) {
    int lane = laneid();
    const bf16* p = Wt + (size_t)(n0 + (lane & 15)) * ld + k0 + ((lane >> 4) << 3);
    v8bf lo = *(const v8bf*)(p);
    v8bf hi = *(const v8bf*)(p + 16);
    v16bf r;
#pragma unroll
    for (int e = 0; e < 8; ++e) { r[e] = lo[e]; r[e + 8] = hi[e]; }
    return r;
}
__device__ __forceinline__ v8f wmma_bf(v16bf a, v16bf b, v8f c) {
    return __builtin_amdgcn_wmma_f32_16x16x32_bf16(false, a, false, b, (short)0, c, false, false);
}

// ---- prep: zero agg + transpose/convert all weights to bf16 Wt[N][K] ----
__device__ __forceinline__ void tposeK(bf16* dst, const float* src, int Kreal, int Kpad,
                                       int Nn, int tid, int stride) {
    int tot = Nn * Kpad;
    for (int i = tid; i < tot; i += stride) {
        int n = i / Kpad, k = i % Kpad;
        dst[i] = (k < Kreal) ? f2bf(src[(size_t)k * Nn + n]) : f2bf(0.0f);
    }
}
__global__ void k_prep(const float* ne_w1, const float* ne_w2, const float* ee_w1, const float* ee_w2,
                       const float* mp_ew, const float* mp_nw, const float* d_w1, const float* d_w2,
                       bf16* ne1t, bf16* ne2t, bf16* ee1t, bf16* ee2t,
                       bf16* ewt, bf16* nwt, bf16* dw1t, bf16* dw2t, float* agg) {
    int tid = blockIdx.x * blockDim.x + threadIdx.x;
    int stride = gridDim.x * blockDim.x;
    for (int i = tid; i < NN * HD; i += stride) agg[i] = 0.0f;
    tposeK(ne1t, ne_w1, 10, 32, HD, tid, stride);
    tposeK(ne2t, ne_w2, HD, HD, HD, tid, stride);
    tposeK(ee1t, ee_w1, 7, 32, HD, tid, stride);
    tposeK(ee2t, ee_w2, HD, HD, HD, tid, stride);
    for (int l = 0; l < NLAY; ++l) {
        tposeK(ewt + (size_t)l * HD * 384, mp_ew + (size_t)l * 384 * HD, 384, 384, HD, tid, stride);
        tposeK(nwt + (size_t)l * HD * 256, mp_nw + (size_t)l * 256 * HD, 256, 256, HD, tid, stride);
    }
    tposeK(dw1t, d_w1, HD, HD, HD, tid, stride);
    tposeK(dw2t, d_w2, HD, HD, 64, tid, stride);
}

// ---- node encoder: h = relu(x@w1+b1)@w2+b2 ; writes h (f32) and h_bf ----
__global__ void __launch_bounds__(128) k_node_encode(const float* __restrict__ x,
                                                     const float* __restrict__ b1,
                                                     const float* __restrict__ b2,
                                                     const bf16* __restrict__ w1t,
                                                     const bf16* __restrict__ w2t,
                                                     float* __restrict__ hf32,
                                                     bf16* __restrict__ hbf) {
    const int wave = threadIdx.x >> 5;
    const int tile = blockIdx.x * WPB + wave;
    if (tile >= NN / 16) return;
    const int lane = laneid(), m = lane & 15, cm = (lane >> 4) * 8;
    __shared__ bf16 sm[WPB][16 * HD];
    bf16* S = sm[wave];
    const int row = tile * 16 + m;

    v16bf a1 = load_a_pad(x + (size_t)row * 10, 10);
    v8f acc[8];
#pragma unroll
    for (int nt = 0; nt < 8; ++nt) { v8f z = {}; acc[nt] = wmma_bf(a1, load_b(w1t, 32, nt * 16, 0), z); }
#pragma unroll
    for (int nt = 0; nt < 8; ++nt) {
        float bia = b1[nt * 16 + m];
#pragma unroll
        for (int r = 0; r < 8; ++r) {
            float v = acc[nt][r] + bia;
            S[(cm + r) * HD + nt * 16 + m] = f2bf(v > 0.0f ? v : 0.0f);
        }
    }
    asm volatile("s_wait_dscnt 0" ::: "memory");
    v8f a2[8];
#pragma unroll
    for (int nt = 0; nt < 8; ++nt) { v8f z = {}; a2[nt] = z; }
#pragma unroll
    for (int ks = 0; ks < 4; ++ks) {
        v16bf a = load_a_g(S + m * HD + ks * 32);
#pragma unroll
        for (int nt = 0; nt < 8; ++nt) a2[nt] = wmma_bf(a, load_b(w2t, HD, nt * 16, ks * 32), a2[nt]);
    }
#pragma unroll
    for (int nt = 0; nt < 8; ++nt) {
        float bia = b2[nt * 16 + m];
#pragma unroll
        for (int r = 0; r < 8; ++r) {
            float v = a2[nt][r] + bia;
            size_t idx = (size_t)(tile * 16 + cm + r) * HD + nt * 16 + m;
            hf32[idx] = v;
            hbf[idx] = f2bf(v);
        }
    }
}

// ---- edge encoder: e_bf = relu(ea@w1+b1)@w2+b2 (bf16 out) ----
__global__ void __launch_bounds__(128) k_edge_encode(const float* __restrict__ ea,
                                                     const float* __restrict__ b1,
                                                     const float* __restrict__ b2,
                                                     const bf16* __restrict__ w1t,
                                                     const bf16* __restrict__ w2t,
                                                     bf16* __restrict__ ebf) {
    const int wave = threadIdx.x >> 5;
    const int tile = blockIdx.x * WPB + wave;
    if (tile >= NE / 16) return;
    const int lane = laneid(), m = lane & 15, cm = (lane >> 4) * 8;
    __shared__ bf16 sm[WPB][16 * HD];
    bf16* S = sm[wave];
    const int row = tile * 16 + m;

    v16bf a1 = load_a_pad(ea + (size_t)row * 7, 7);
    v8f acc[8];
#pragma unroll
    for (int nt = 0; nt < 8; ++nt) { v8f z = {}; acc[nt] = wmma_bf(a1, load_b(w1t, 32, nt * 16, 0), z); }
#pragma unroll
    for (int nt = 0; nt < 8; ++nt) {
        float bia = b1[nt * 16 + m];
#pragma unroll
        for (int r = 0; r < 8; ++r) {
            float v = acc[nt][r] + bia;
            S[(cm + r) * HD + nt * 16 + m] = f2bf(v > 0.0f ? v : 0.0f);
        }
    }
    asm volatile("s_wait_dscnt 0" ::: "memory");
    v8f a2[8];
#pragma unroll
    for (int nt = 0; nt < 8; ++nt) { v8f z = {}; a2[nt] = z; }
#pragma unroll
    for (int ks = 0; ks < 4; ++ks) {
        v16bf a = load_a_g(S + m * HD + ks * 32);
#pragma unroll
        for (int nt = 0; nt < 8; ++nt) a2[nt] = wmma_bf(a, load_b(w2t, HD, nt * 16, ks * 32), a2[nt]);
    }
#pragma unroll
    for (int nt = 0; nt < 8; ++nt) {
        float bia = b2[nt * 16 + m];
#pragma unroll
        for (int r = 0; r < 8; ++r)
            ebf[(size_t)(tile * 16 + cm + r) * HD + nt * 16 + m] = f2bf(a2[nt][r] + bia);
    }
}

// ---- edge MP: msg = relu([h[src]|h[dst]|e] @ ew + eb); agg[dst] += msg ----
__global__ void __launch_bounds__(128) k_edge_mp(const bf16* __restrict__ hbf,
                                                 const bf16* __restrict__ ebf,
                                                 float* __restrict__ agg,
                                                 const int* __restrict__ srcI,
                                                 const int* __restrict__ dstI,
                                                 const bf16* __restrict__ ewt,
                                                 const float* __restrict__ eb) {
    const int wave = threadIdx.x >> 5;
    const int tile = blockIdx.x * WPB + wave;
    if (tile >= NE / 16) return;
    const int lane = laneid(), m = lane & 15, cm = (lane >> 4) * 8;
    const int e0 = tile * 16;
    const int es = e0 + m;
    const int srow = srcI[es];
    const int drow = dstI[es];
    const bf16* aS = hbf + (size_t)srow * HD;
    const bf16* aD = hbf + (size_t)drow * HD;
    const bf16* aE = ebf + (size_t)es * HD;

    v8f acc[8];
#pragma unroll
    for (int nt = 0; nt < 8; ++nt) { v8f z = {}; acc[nt] = z; }
#pragma unroll
    for (int ks = 0; ks < 12; ++ks) {
        const bf16* ap = (ks < 4) ? (aS + ks * 32) : ((ks < 8) ? (aD + (ks - 4) * 32) : (aE + (ks - 8) * 32));
        v16bf a = load_a_g(ap);
#pragma unroll
        for (int nt = 0; nt < 8; ++nt) acc[nt] = wmma_bf(a, load_b(ewt, 384, nt * 16, ks * 32), acc[nt]);
    }
    int drows[8];
#pragma unroll
    for (int r = 0; r < 8; ++r) drows[r] = dstI[e0 + cm + r];
#pragma unroll
    for (int nt = 0; nt < 8; ++nt) {
        float bia = eb[nt * 16 + m];
#pragma unroll
        for (int r = 0; r < 8; ++r) {
            float v = acc[nt][r] + bia;
            v = v > 0.0f ? v : 0.0f;
            atomicAdd(&agg[(size_t)drows[r] * HD + nt * 16 + m], v);
        }
    }
}

// ---- node MP: h += relu([h|agg] @ nw + nb); re-zero agg for next layer ----
__global__ void __launch_bounds__(128) k_node_mp(float* __restrict__ hf32,
                                                 bf16* __restrict__ hbf,
                                                 float* __restrict__ agg,
                                                 const bf16* __restrict__ nwt,
                                                 const float* __restrict__ nb) {
    const int wave = threadIdx.x >> 5;
    const int tile = blockIdx.x * WPB + wave;
    if (tile >= NN / 16) return;
    const int lane = laneid(), m = lane & 15, cm = (lane >> 4) * 8;
    const int row = tile * 16 + m;

    v8f acc[8];
#pragma unroll
    for (int nt = 0; nt < 8; ++nt) { v8f z = {}; acc[nt] = z; }
#pragma unroll
    for (int ks = 0; ks < 8; ++ks) {
        v16bf a = (ks < 4) ? load_a_g(hbf + (size_t)row * HD + ks * 32)
                           : load_a_f32(agg + (size_t)row * HD + (ks - 4) * 32);
#pragma unroll
        for (int nt = 0; nt < 8; ++nt) acc[nt] = wmma_bf(a, load_b(nwt, 256, nt * 16, ks * 32), acc[nt]);
    }
#pragma unroll
    for (int nt = 0; nt < 8; ++nt) {
        float bia = nb[nt * 16 + m];
#pragma unroll
        for (int r = 0; r < 8; ++r) {
            size_t idx = (size_t)(tile * 16 + cm + r) * HD + nt * 16 + m;
            float u = acc[nt][r] + bia;
            u = u > 0.0f ? u : 0.0f;
            float hn = hf32[idx] + u;
            hf32[idx] = hn;
            hbf[idx] = f2bf(hn);
            agg[idx] = 0.0f;   // reads of agg happened in K-loop above
        }
    }
}

// ---- decoder: relu(h@w1+b1) -> relu(@w2+b2) -> @w3+b3, then BC mask ----
__global__ void __launch_bounds__(128) k_decode(const bf16* __restrict__ hbf,
                                                const float* __restrict__ b1,
                                                const float* __restrict__ b2,
                                                const bf16* __restrict__ w1t,
                                                const bf16* __restrict__ w2t,
                                                const float* __restrict__ w3,
                                                const float* __restrict__ b3,
                                                const float* __restrict__ bc_disp,
                                                const float* __restrict__ bc_rot,
                                                float* __restrict__ out) {
    const int wave = threadIdx.x >> 5;
    const int tile = blockIdx.x * WPB + wave;
    if (tile >= NN / 16) return;
    const int lane = laneid(), m = lane & 15, cm = (lane >> 4) * 8;
    __shared__ bf16 z1s[WPB][16 * HD];
    __shared__ float z2s[WPB][16 * 64];
    bf16* Z1 = z1s[wave];
    float* Z2 = z2s[wave];
    const int row = tile * 16 + m;

    v8f acc[8];
#pragma unroll
    for (int nt = 0; nt < 8; ++nt) { v8f z = {}; acc[nt] = z; }
#pragma unroll
    for (int ks = 0; ks < 4; ++ks) {
        v16bf a = load_a_g(hbf + (size_t)row * HD + ks * 32);
#pragma unroll
        for (int nt = 0; nt < 8; ++nt) acc[nt] = wmma_bf(a, load_b(w1t, HD, nt * 16, ks * 32), acc[nt]);
    }
#pragma unroll
    for (int nt = 0; nt < 8; ++nt) {
        float bia = b1[nt * 16 + m];
#pragma unroll
        for (int r = 0; r < 8; ++r) {
            float v = acc[nt][r] + bia;
            Z1[(cm + r) * HD + nt * 16 + m] = f2bf(v > 0.0f ? v : 0.0f);
        }
    }
    asm volatile("s_wait_dscnt 0" ::: "memory");
    v8f a2[4];
#pragma unroll
    for (int nt = 0; nt < 4; ++nt) { v8f z = {}; a2[nt] = z; }
#pragma unroll
    for (int ks = 0; ks < 4; ++ks) {
        v16bf a = load_a_g(Z1 + m * HD + ks * 32);
#pragma unroll
        for (int nt = 0; nt < 4; ++nt) a2[nt] = wmma_bf(a, load_b(w2t, HD, nt * 16, ks * 32), a2[nt]);
    }
#pragma unroll
    for (int nt = 0; nt < 4; ++nt) {
        float bia = b2[nt * 16 + m];
#pragma unroll
        for (int r = 0; r < 8; ++r) {
            float v = a2[nt][r] + bia;
            Z2[(cm + r) * 64 + nt * 16 + m] = v > 0.0f ? v : 0.0f;
        }
    }
    asm volatile("s_wait_dscnt 0" ::: "memory");
    if (lane < 16) {
        int node = tile * 16 + lane;
        float s0 = b3[0], s1 = b3[1], s2 = b3[2];
#pragma unroll 8
        for (int k = 0; k < 64; ++k) {
            float zz = Z2[lane * 64 + k];
            s0 += zz * w3[k * 3 + 0];
            s1 += zz * w3[k * 3 + 1];
            s2 += zz * w3[k * 3 + 2];
        }
        float md = 1.0f - bc_disp[node];
        float mr = 1.0f - bc_rot[node];
        out[node * 3 + 0] = s0 * md;
        out[node * 3 + 1] = s1 * md;
        out[node * 3 + 2] = s2 * mr;
    }
}

extern "C" void kernel_launch(void* const* d_in, const int* in_sizes, int n_in,
                              void* d_out, int out_size, void* d_ws, size_t ws_size,
                              hipStream_t stream) {
    const float* x       = (const float*)d_in[0];
    const float* ea      = (const float*)d_in[1];
    const int*   eidx    = (const int*)d_in[2];
    const float* bc_disp = (const float*)d_in[3];
    const float* bc_rot  = (const float*)d_in[4];
    const float* ne_w1 = (const float*)d_in[5];
    const float* ne_b1 = (const float*)d_in[6];
    const float* ne_w2 = (const float*)d_in[7];
    const float* ne_b2 = (const float*)d_in[8];
    const float* ee_w1 = (const float*)d_in[9];
    const float* ee_b1 = (const float*)d_in[10];
    const float* ee_w2 = (const float*)d_in[11];
    const float* ee_b2 = (const float*)d_in[12];
    const float* mp_ew = (const float*)d_in[13];
    const float* mp_eb = (const float*)d_in[14];
    const float* mp_nw = (const float*)d_in[15];
    const float* mp_nb = (const float*)d_in[16];
    const float* d_w1  = (const float*)d_in[17];
    const float* d_b1  = (const float*)d_in[18];
    const float* d_w2  = (const float*)d_in[19];
    const float* d_b2  = (const float*)d_in[20];
    const float* d_w3  = (const float*)d_in[21];
    const float* d_b3  = (const float*)d_in[22];

    char* ws = (char*)d_ws;
    size_t off = 0;
    float* hf32 = (float*)(ws + off); off += (size_t)NN * HD * 4;
    bf16*  hbf  = (bf16*)(ws + off);  off += (size_t)NN * HD * 2;
    float* agg  = (float*)(ws + off); off += (size_t)NN * HD * 4;
    bf16*  ebf  = (bf16*)(ws + off);  off += (size_t)NE * HD * 2;
    bf16* ne1t = (bf16*)(ws + off); off += 128 * 32 * 2;
    bf16* ne2t = (bf16*)(ws + off); off += 128 * 128 * 2;
    bf16* ee1t = (bf16*)(ws + off); off += 128 * 32 * 2;
    bf16* ee2t = (bf16*)(ws + off); off += 128 * 128 * 2;
    bf16* ewt  = (bf16*)(ws + off); off += (size_t)NLAY * 128 * 384 * 2;
    bf16* nwt  = (bf16*)(ws + off); off += (size_t)NLAY * 128 * 256 * 2;
    bf16* dw1t = (bf16*)(ws + off); off += 128 * 128 * 2;
    bf16* dw2t = (bf16*)(ws + off); off += 64 * 128 * 2;

    const int* srcI = eidx;
    const int* dstI = eidx + NE;

    const int NTILES = NN / 16;   // 3125
    const int ETILES = NE / 16;   // 50000
    const int nBlocks = (NTILES + WPB - 1) / WPB;
    const int eBlocks = (ETILES + WPB - 1) / WPB;

    k_prep<<<2048, 256, 0, stream>>>(ne_w1, ne_w2, ee_w1, ee_w2, mp_ew, mp_nw, d_w1, d_w2,
                                     ne1t, ne2t, ee1t, ee2t, ewt, nwt, dw1t, dw2t, agg);
    k_node_encode<<<nBlocks, 128, 0, stream>>>(x, ne_b1, ne_b2, ne1t, ne2t, hf32, hbf);
    k_edge_encode<<<eBlocks, 128, 0, stream>>>(ea, ee_b1, ee_b2, ee1t, ee2t, ebf);
    for (int l = 0; l < NLAY; ++l) {
        k_edge_mp<<<eBlocks, 128, 0, stream>>>(hbf, ebf, agg, srcI, dstI,
                                               ewt + (size_t)l * 128 * 384, mp_eb + (size_t)l * HD);
        k_node_mp<<<nBlocks, 128, 0, stream>>>(hf32, hbf, agg,
                                               nwt + (size_t)l * 128 * 256, mp_nb + (size_t)l * HD);
    }
    k_decode<<<nBlocks, 128, 0, stream>>>(hbf, d_b1, d_b2, dw1t, dw2t, d_w3, d_b3,
                                          bc_disp, bc_rot, (float*)d_out);
}
